// AlexNet_Q_40329742910117
// MI455X (gfx1250) — matrix operand order repaired and checked by reading, not compiled, not run on hardware
//
#include <hip/hip_runtime.h>
#include <cstdint>
#include <cstddef>

// ---------------------------------------------------------------------------
// Types for CDNA5 WMMA (wave32)
// ---------------------------------------------------------------------------
typedef __attribute__((ext_vector_type(16))) _Float16 v16h;
typedef __attribute__((ext_vector_type(8)))  _Float16 v8h;
typedef __attribute__((ext_vector_type(8)))  float    v8f;

#define BM 128
#define BN 128
#define BK 32
#define BKP 40            // LDS row stride in halves (80B: bank-conflict-free, 16B aligned)
#define GEMM_THREADS 256  // 8 wave32 waves

// ---------------------------------------------------------------------------
// CDNA5 async global->LDS copy (ASYNCcnt-tracked), per ISA 15.18.3 / 08_async
// ---------------------------------------------------------------------------
__device__ __forceinline__ uint32_t lds_off(const void* p) {
  // Generic (flat) pointers to LDS carry the LDS byte offset in the low 32 bits.
  return (uint32_t)(uintptr_t)p;
}

__device__ __forceinline__ void async_copy_b128(uint32_t lds, uint64_t gaddr) {
  asm volatile("global_load_async_to_lds_b128 %0, %1, off"
               :: "v"(lds), "v"(gaddr) : "memory");
}

__device__ __forceinline__ void wait_async_0() {
  asm volatile("s_wait_asynccnt 0" ::: "memory");
}

// ---------------------------------------------------------------------------
// LDS fragment loads matching CDNA5 WMMA VGPR layouts (wave32)
// ---------------------------------------------------------------------------
__device__ __forceinline__ v16h load_a_frag(const _Float16* T, int row0, int lane) {
  // A 16x32 f16: lanes 0-15 -> rows 0..15, K {0..7, 16..23}; lanes 16-31 -> K {8..15, 24..31}
  const _Float16* p = T + (size_t)(row0 + (lane & 15)) * BKP + ((lane >> 4) << 3);
  v8h lo = *(const v8h*)p;
  v8h hi = *(const v8h*)(p + 16);
  return __builtin_shufflevector(lo, hi, 0,1,2,3,4,5,6,7,8,9,10,11,12,13,14,15);
}

__device__ __forceinline__ v16h load_b_frag(const _Float16* T, int col0, int lane) {
  // B 32x16 f16: lanes 0-15 -> col=lane, K {0..7, 16..23}; lanes 16-31 -> col=lane-16, K {8..15, 24..31}
  const _Float16* p = T + (size_t)(col0 + (lane & 15)) * BKP + ((lane >> 4) << 3);
  v8h lo = *(const v8h*)p;
  v8h hi = *(const v8h*)(p + 16);
  return __builtin_shufflevector(lo, hi, 0,1,2,3,4,5,6,7,8,9,10,11,12,13,14,15);
}

// Issue one 128x32 A tile + 128x32 B tile of async b128 copies (4 per thread).
__device__ __forceinline__ void issue_tile(
    const _Float16* __restrict__ A, const _Float16* __restrict__ Bw,
    int M, int N, int KP, int blockM, int blockN, int kb,
    _Float16* As, _Float16* Bs, int tid)
{
  const size_t kbase = (size_t)kb * BK;
#pragma unroll
  for (int i = 0; i < 2; ++i) {
    int c = tid + i * GEMM_THREADS;
    int row = c >> 2, cc = (c & 3) * 8;
    int rg = blockM + row; if (rg > M - 1) rg = M - 1;   // clamp: partial M tiles
    async_copy_b128(lds_off(As + (size_t)row * BKP + cc),
                    (uint64_t)(uintptr_t)(A + (size_t)rg * KP + kbase + cc));
  }
#pragma unroll
  for (int i = 0; i < 2; ++i) {
    int c = tid + i * GEMM_THREADS;
    int row = c >> 2, cc = (c & 3) * 8;
    int rg = blockN + row; if (rg > N - 1) rg = N - 1;   // clamp: partial N tiles
    async_copy_b128(lds_off(Bs + (size_t)row * BKP + cc),
                    (uint64_t)(uintptr_t)(Bw + (size_t)rg * KP + kbase + cc));
  }
}

// ---------------------------------------------------------------------------
// Tiled WMMA GEMM: C[M,N] = A[M,KP] * Bw[N,KP]^T  (f16 in, f32 out)
// Epilogue: x = acc + bias[n]; optional x = x*bnsc[n]+bnsh[n]; optional ReLU.
// KP must be a multiple of 32 (callers zero-pad K).
// ---------------------------------------------------------------------------
__global__ __launch_bounds__(GEMM_THREADS)
void gemm_wmma_f16(const _Float16* __restrict__ A, const _Float16* __restrict__ Bw,
                   float* __restrict__ C, int M, int N, int KP,
                   const float* __restrict__ bias, const float* __restrict__ bnsc,
                   const float* __restrict__ bnsh, int relu)
{
  __shared__ _Float16 As[2][BM * BKP];   // 2 x 10 KB
  __shared__ _Float16 Bs[2][BN * BKP];   // 2 x 10 KB
  const int tid    = threadIdx.x;
  const int lane   = tid & 31;
  const int wv     = tid >> 5;
  const int waveM  = (wv >> 2) * 64;     // 2 waves in M
  const int waveN  = (wv & 3) * 32;      // 4 waves in N
  const int blockM = blockIdx.y * BM;
  const int blockN = blockIdx.x * BN;
  const int nkb    = KP / BK;

  const v8f zero = {0.f, 0.f, 0.f, 0.f, 0.f, 0.f, 0.f, 0.f};
  v8f acc[4][2];
#pragma unroll
  for (int mi = 0; mi < 4; ++mi)
#pragma unroll
    for (int ni = 0; ni < 2; ++ni) acc[mi][ni] = zero;

  issue_tile(A, Bw, M, N, KP, blockM, blockN, 0, As[0], Bs[0], tid);

  int buf = 0;
  for (int kb = 0; kb < nkb; ++kb) {
    wait_async_0();        // own async copies of tile kb complete
    __syncthreads();       // everyone's copies complete; prev buffer free
    if (kb + 1 < nkb)
      issue_tile(A, Bw, M, N, KP, blockM, blockN, kb + 1, As[buf ^ 1], Bs[buf ^ 1], tid);
    if (kb + 2 < nkb) {    // warm L2 for tile kb+2 (global_prefetch_b8)
      int row = tid >> 2, cc = (tid & 3) * 8;
      int rg = blockM + row; if (rg > M - 1) rg = M - 1;
      __builtin_prefetch(A + (size_t)rg * KP + (size_t)(kb + 2) * BK + cc, 0, 1);
    }

    const _Float16* At = As[buf];
    const _Float16* Bt = Bs[buf];
    v16h af[4], bf[2];
#pragma unroll
    for (int mi = 0; mi < 4; ++mi) af[mi] = load_a_frag(At, waveM + mi * 16, lane);
#pragma unroll
    for (int ni = 0; ni < 2; ++ni) bf[ni] = load_b_frag(Bt, waveN + ni * 16, lane);
#pragma unroll
    for (int mi = 0; mi < 4; ++mi)
#pragma unroll
      for (int ni = 0; ni < 2; ++ni)
        acc[mi][ni] = __builtin_amdgcn_wmma_f32_16x16x32_f16(
            false, af[mi], false, bf[ni], (short)0, acc[mi][ni], false, false);
    buf ^= 1;
  }

  // C/D layout: VGPR v -> row v (+8 for lanes 16-31), col = lane & 15.
  const int rsel = (lane >> 4) * 8;
  const int csel = lane & 15;
#pragma unroll
  for (int mi = 0; mi < 4; ++mi)
#pragma unroll
    for (int ni = 0; ni < 2; ++ni)
#pragma unroll
      for (int v = 0; v < 8; ++v) {
        int r = blockM + waveM + mi * 16 + rsel + v;
        int c = blockN + waveN + ni * 16 + csel;
        if (r < M && c < N) {
          float xv = acc[mi][ni][v] + bias[c];
          if (bnsc) xv = xv * bnsc[c] + bnsh[c];
          if (relu) xv = fmaxf(xv, 0.f);
          C[(size_t)r * N + c] = xv;
        }
      }
}

// ---------------------------------------------------------------------------
// Weight quantization: E = sqrt(mean(w^2)); wq = sign(w)*E -> f16, K zero-padded
// ---------------------------------------------------------------------------
__global__ void sumsq_kernel(const float* __restrict__ w, float* __restrict__ acc, size_t n) {
  __shared__ float red[256];
  size_t stride = (size_t)gridDim.x * blockDim.x;
  float s = 0.f;
  for (size_t i = blockIdx.x * (size_t)blockDim.x + threadIdx.x; i < n; i += stride) {
    float v = w[i]; s += v * v;
  }
  red[threadIdx.x] = s;
  __syncthreads();
  for (int o = 128; o > 0; o >>= 1) {
    if ((int)threadIdx.x < o) red[threadIdx.x] += red[threadIdx.x + o];
    __syncthreads();
  }
  if (threadIdx.x == 0) atomicAdd(acc, red[0]);
}

__global__ void quant_sign_f16_kernel(const float* __restrict__ w, const float* __restrict__ acc,
                                      _Float16* __restrict__ out, int K, int KP,
                                      size_t total, float invn) {
  size_t i = blockIdx.x * (size_t)blockDim.x + threadIdx.x;
  if (i >= total) return;
  int k = (int)(i % KP);
  size_t n = i / KP;
  float v = 0.f;
  if (k < K) {
    float E = sqrtf(*acc * invn);
    float x = w[n * (size_t)K + k];
    float s = (x > 0.f) ? 1.f : ((x < 0.f) ? -1.f : 0.f);  // jnp.sign semantics
    v = s * E;
  }
  out[i] = (_Float16)v;
}

__global__ void convert_pad_f16_kernel(const float* __restrict__ w, _Float16* __restrict__ out,
                                       int K, int KP, size_t total) {
  size_t i = blockIdx.x * (size_t)blockDim.x + threadIdx.x;
  if (i >= total) return;
  int k = (int)(i % KP);
  size_t n = i / KP;
  out[i] = (_Float16)((k < K) ? w[n * (size_t)K + k] : 0.f);
}

// ---------------------------------------------------------------------------
// BatchNorm fold: sc = g/sqrt(v+eps), sh = be - m*sc
// ---------------------------------------------------------------------------
__global__ void bn_prep_kernel(const float* g, const float* be, const float* m,
                               const float* v, float* sc, float* sh, int C) {
  int c = blockIdx.x * blockDim.x + threadIdx.x;
  if (c >= C) return;
  float inv = g[c] * rsqrtf(v[c] + 1e-5f);
  sc[c] = inv;
  sh[c] = be[c] - m[c] * inv;
}

// ---------------------------------------------------------------------------
// im2col -> f16, K zero-padded to KP. k order = c*KH*KW + kh*KW + kw (NCHW conv).
// src layout: nhwc ? [N,H,W,C] : [N,C,H,W].  Rows m0..m0+mc-1 of the patch matrix.
// ---------------------------------------------------------------------------
__global__ void im2col_f16_kernel(const float* __restrict__ src, _Float16* __restrict__ dst,
                                  int m0, int C, int H, int W, int KH, int KW,
                                  int st, int pd, int OH, int OW, int K, int KP,
                                  int nhwc, size_t total) {
  size_t i = blockIdx.x * (size_t)blockDim.x + threadIdx.x;
  if (i >= total) return;
  int kp = (int)(i % KP);
  int m  = m0 + (int)(i / KP);
  float val = 0.f;
  if (kp < K) {
    int ow = m % OW; int t = m / OW;
    int oh = t % OH; int n = t / OH;
    int kw = kp % KW; int t2 = kp / KW;
    int kh = t2 % KH; int c  = t2 / KH;
    int ih = oh * st - pd + kh;
    int iw = ow * st - pd + kw;
    if (ih >= 0 && ih < H && iw >= 0 && iw < W) {
      size_t idx = nhwc ? (((size_t)(n * H + ih) * W + iw) * C + c)
                        : (((size_t)(n * C + c) * H + ih) * W + iw);
      val = src[idx];
    }
  }
  dst[i] = (_Float16)val;
}

// ---------------------------------------------------------------------------
// 3x3 stride-2 VALID maxpool on NHWC f32 (window always in bounds for VALID).
// ---------------------------------------------------------------------------
__global__ void maxpool3s2_nhwc_kernel(const float* __restrict__ in, float* __restrict__ out,
                                       int C, int H, int W, int OH, int OW, size_t total) {
  size_t i = blockIdx.x * (size_t)blockDim.x + threadIdx.x;
  if (i >= total) return;
  int c  = (int)(i % C); size_t t = i / C;
  int ow = (int)(t % OW); t /= OW;
  int oh = (int)(t % OH);
  int n  = (int)(t / OH);
  float m = -3.402823466e38f;
#pragma unroll
  for (int dy = 0; dy < 3; ++dy)
#pragma unroll
    for (int dx = 0; dx < 3; ++dx) {
      int ih = oh * 2 + dy, iw = ow * 2 + dx;
      m = fmaxf(m, in[((size_t)(n * H + ih) * W + iw) * C + c]);
    }
  out[i] = m;
}

// ---------------------------------------------------------------------------
// Launcher: full quantized-AlexNet forward as a sequence of kernels on `stream`
// ---------------------------------------------------------------------------
extern "C" void kernel_launch(void* const* d_in, const int* in_sizes, int n_in,
                              void* d_out, int out_size, void* d_ws, size_t ws_size,
                              hipStream_t stream) {
  (void)in_sizes; (void)n_in; (void)out_size; (void)ws_size;

  const float* x   = (const float*)d_in[0];
  const float* w1  = (const float*)d_in[1];
  const float* b1  = (const float*)d_in[2];
  const float* g1  = (const float*)d_in[3];
  const float* be1 = (const float*)d_in[4];
  const float* m1  = (const float*)d_in[5];
  const float* v1  = (const float*)d_in[6];
  const float* w2  = (const float*)d_in[7];
  const float* b2  = (const float*)d_in[8];
  const float* g2  = (const float*)d_in[9];
  const float* be2 = (const float*)d_in[10];
  const float* m2  = (const float*)d_in[11];
  const float* v2  = (const float*)d_in[12];
  const float* w3  = (const float*)d_in[13];
  const float* b3  = (const float*)d_in[14];
  const float* w4  = (const float*)d_in[15];
  const float* b4  = (const float*)d_in[16];
  const float* w5  = (const float*)d_in[17];
  const float* b5  = (const float*)d_in[18];
  const float* fw1 = (const float*)d_in[19];
  const float* fb1 = (const float*)d_in[20];
  const float* fw2 = (const float*)d_in[21];
  const float* fb2 = (const float*)d_in[22];
  const float* fw3 = (const float*)d_in[23];
  const float* fb3 = (const float*)d_in[24];
  float* out = (float*)d_out;

  // Workspace carve-up (~306 MB used).
  char* ws = (char*)d_ws;
  float* qacc  = (float*)(ws + 0);          // 1 float (reduction accumulator)
  float* bn1sc = (float*)(ws + 256);
  float* bn1sh = bn1sc + 96;
  float* bn2sc = bn1sh + 96;
  float* bn2sh = bn2sc + 256;
  size_t off = 4096;
  _Float16* Abuf = (_Float16*)(ws + off); off += (size_t)125 << 20;  // im2col tiles
  _Float16* Bbuf = (_Float16*)(ws + off); off += (size_t)80  << 20;  // f16 weights
  float*    actX = (float*)(ws + off);    off += (size_t)80  << 20;  // NHWC activations
  float*    actY = (float*)(ws + off);    off += (size_t)20  << 20;

  auto quant = [&](const float* w, int N, int K, int KP) {
    hipMemsetAsync(qacc, 0, sizeof(float), stream);
    size_t n = (size_t)N * K;
    int blocks = (int)((n + 255) / 256); if (blocks > 1024) blocks = 1024;
    sumsq_kernel<<<blocks, 256, 0, stream>>>(w, qacc, n);
    size_t tot = (size_t)N * KP;
    quant_sign_f16_kernel<<<(unsigned)((tot + 255) / 256), 256, 0, stream>>>(
        w, qacc, Bbuf, K, KP, tot, 1.0f / (float)n);
  };
  auto convertW = [&](const float* w, int N, int K, int KP) {
    size_t tot = (size_t)N * KP;
    convert_pad_f16_kernel<<<(unsigned)((tot + 255) / 256), 256, 0, stream>>>(
        w, Bbuf, K, KP, tot);
  };
  // conv / FC via im2col + WMMA GEMM, chunked over M to bound Abuf.
  auto conv = [&](const float* act, int nhwc, int Bn, int Cc, int H, int W,
                  int KH, int KW, int st, int pd, int OC, int KP, float* outp,
                  const float* bias, const float* bsc, const float* bsh, int relu) {
    int K  = Cc * KH * KW;
    int OH = (H + 2 * pd - KH) / st + 1;
    int OW = (W + 2 * pd - KW) / st + 1;
    int M  = Bn * OH * OW;
    const int MCHUNK = 24576;
    for (int m0 = 0; m0 < M; m0 += MCHUNK) {
      int mc = M - m0; if (mc > MCHUNK) mc = MCHUNK;
      size_t tot = (size_t)mc * KP;
      im2col_f16_kernel<<<(unsigned)((tot + 255) / 256), 256, 0, stream>>>(
          act, Abuf, m0, Cc, H, W, KH, KW, st, pd, OH, OW, K, KP, nhwc, tot);
      dim3 grid((OC + BN - 1) / BN, (mc + BM - 1) / BM);
      gemm_wmma_f16<<<grid, GEMM_THREADS, 0, stream>>>(
          Abuf, Bbuf, outp + (size_t)m0 * OC, mc, OC, KP, bias, bsc, bsh, relu);
    }
  };
  auto pool = [&](const float* in, float* outp, int Bn, int Cc, int H, int W) {
    int OH = (H - 3) / 2 + 1, OW = (W - 3) / 2 + 1;
    size_t tot = (size_t)Bn * OH * OW * Cc;
    maxpool3s2_nhwc_kernel<<<(unsigned)((tot + 255) / 256), 256, 0, stream>>>(
        in, outp, Cc, H, W, OH, OW, tot);
  };

  // Fold BN params once.
  bn_prep_kernel<<<1, 96, 0, stream>>>(g1, be1, m1, v1, bn1sc, bn1sh, 96);
  bn_prep_kernel<<<1, 256, 0, stream>>>(g2, be2, m2, v2, bn2sc, bn2sh, 256);

  // conv1 (fp weights) : NCHW input -> NHWC h1 [64,55,55,96]
  convertW(w1, 96, 363, 384);
  conv(x, 0, 64, 3, 224, 224, 11, 11, 4, 2, 96, 384, actX, b1, bn1sc, bn1sh, 1);
  pool(actX, actY, 64, 96, 55, 55);                       // -> [64,27,27,96]

  // conv2 (1-bit) : -> [64,27,27,256]
  quant(w2, 256, 2400, 2400);
  conv(actY, 1, 64, 96, 27, 27, 5, 5, 1, 2, 256, 2400, actX, b2, bn2sc, bn2sh, 1);
  pool(actX, actY, 64, 256, 27, 27);                      // -> [64,13,13,256]

  // conv3/4/5 (1-bit)
  quant(w3, 384, 2304, 2304);
  conv(actY, 1, 64, 256, 13, 13, 3, 3, 1, 1, 384, 2304, actX, b3, nullptr, nullptr, 1);
  quant(w4, 384, 3456, 3456);
  conv(actX, 1, 64, 384, 13, 13, 3, 3, 1, 1, 384, 3456, actY, b4, nullptr, nullptr, 1);
  quant(w5, 256, 3456, 3456);
  conv(actY, 1, 64, 384, 13, 13, 3, 3, 1, 1, 256, 3456, actX, b5, nullptr, nullptr, 1);
  pool(actX, actY, 64, 256, 13, 13);                      // -> [64,6,6,256]

  // fc1 as 6x6 conv (k order matches NCHW flatten that fw1 expects)
  quant(fw1, 4096, 9216, 9216);
  conv(actY, 1, 64, 256, 6, 6, 6, 6, 1, 0, 4096, 9216, actX, fb1, nullptr, nullptr, 1);
  // fc2 as 1x1 conv
  quant(fw2, 4096, 4096, 4096);
  conv(actX, 1, 64, 4096, 1, 1, 1, 1, 1, 0, 4096, 4096, actY, fb2, nullptr, nullptr, 1);
  // fc3 (fp weights, no ReLU) -> d_out [64,1000]
  convertW(fw3, 1000, 4096, 4096);
  conv(actY, 1, 64, 4096, 1, 1, 1, 1, 1, 0, 1000, 4096, out, fb3, nullptr, nullptr, 0);
}
